// WeightShareSelfAttention_28956669509876
// MI455X (gfx1250) — matrix operand order repaired and checked by reading, not compiled, not run on hardware
//
#include <hip/hip_runtime.h>
#include <hip/hip_bf16.h>
#include <stdint.h>

#define BSZ     2
#define NHEAD   16
#define DHEAD   64
#define DMODEL  1024
#define QLEN    2048
#define LOCAL   1000

typedef __attribute__((ext_vector_type(16))) __bf16 v16bf;
typedef __attribute__((ext_vector_type(8)))  float  v8f;

union ABf { v16bf v; uint4 u[2]; };

__device__ __forceinline__ v8f wmma_bf16(v16bf a, v16bf b, v8f c) {
  // D = A(16x32 bf16) x B(32x16 bf16) + C(16x16 f32)
  return __builtin_amdgcn_wmma_f32_16x16x32_bf16(false, a, false, b, (short)0, c, false, false);
}

// ---------------------------------------------------------------------------
// Async staging of a 64(n) x 32(k) bf16 B panel (4 KB) into LDS.
// 128 threads x 2 x b128 = 256 16-byte chunks. ASYNCcnt-tracked DMA path
// (global_load_async_to_lds_b128), no VGPR round trip.
// LDS offset = low 32 bits of the generic pointer (ISA: LDS aperture maps
// addr[31:0] onto the wave's LDS allocation).
// ---------------------------------------------------------------------------
__device__ __forceinline__ void issue_b_async(const __bf16* __restrict__ B, int ldb,
                                              int kk, __bf16* sbuf) {
  const int tid = threadIdx.x;  // 0..127
#pragma unroll
  for (int i = 0; i < 2; ++i) {
    const int c = tid + i * 128;         // chunk 0..255
    const int n = c >> 2;                // B row (n-major), 0..63
    const int cw = c & 3;                // 16B chunk within the 64B row slice
    const __bf16* gp = B + (size_t)n * ldb + kk + cw * 8;
    const unsigned loff = (unsigned)(uintptr_t)(sbuf + c * 8);
    const uint64_t ga = (uint64_t)(uintptr_t)gp;
    asm volatile("global_load_async_to_lds_b128 %0, %1, off"
                 :: "v"(loff), "v"(ga) : "memory");
  }
}

// ---------------------------------------------------------------------------
// Block-cooperative GEMM: 4 waves x 32 rows = 128(M) x 64(N) tile, K=1024.
// B panel double-buffered in LDS via async loads and shared by all 4 waves;
// each wave computes 2(M) x 4(N) = 8 WMMA tiles per k-step (8 WMMA per
// 8 ds_load_b128), A streamed per wave with prefetch.
// ---------------------------------------------------------------------------
__device__ __forceinline__ void gemm_async_128x64(const __bf16* __restrict__ Aw,
                                                  const __bf16* __restrict__ Bb,
                                                  __bf16 (*sB)[64 * 32],
                                                  v8f acc[2][4]) {
  const int lane = threadIdx.x & 31;
  const int half = lane >> 4;
  const int ml   = lane & 15;
  constexpr int S = DMODEL / 32;

  issue_b_async(Bb, DMODEL, 0, sB[0]);
  for (int s = 0; s < S; ++s) {
    if (s + 1 < S) {
      issue_b_async(Bb, DMODEL, (s + 1) * 32, sB[(s + 1) & 1]);
      __builtin_prefetch(Aw + ml * DMODEL + (s + 1) * 32 + half * 8, 0, 3);
      __builtin_prefetch(Aw + (16 + ml) * DMODEL + (s + 1) * 32 + half * 8, 0, 3);
      asm volatile("s_wait_asynccnt 0x2" ::: "memory");   // current panel done
    } else {
      asm volatile("s_wait_asynccnt 0x0" ::: "memory");
    }
    __syncthreads();

    // ---- load phase: 2 A row-tiles (global) + 4 B subtiles (LDS) ----
    ABf a[2], bb[4];
#pragma unroll
    for (int mi = 0; mi < 2; ++mi) {
      const __bf16* ap = Aw + (mi * 16 + ml) * DMODEL + s * 32 + half * 16;
      a[mi].u[0] = *(const uint4*)(ap);
      a[mi].u[1] = *(const uint4*)(ap + 8);
    }
    const __bf16* sb = sB[s & 1];
#pragma unroll
    for (int nt = 0; nt < 4; ++nt) {
      const __bf16* bp = sb + (nt * 16 + ml) * 32 + half * 16;
      bb[nt].u[0] = *(const uint4*)(bp);
      bb[nt].u[1] = *(const uint4*)(bp + 8);
    }
    // ---- compute phase: 8 independent WMMAs ----
#pragma unroll
    for (int mi = 0; mi < 2; ++mi)
#pragma unroll
      for (int nt = 0; nt < 4; ++nt)
        acc[mi][nt] = wmma_bf16(a[mi].v, bb[nt].v, acc[mi][nt]);
    __syncthreads();
  }
}

// ---------------------------------------------------------------------------
// f32 -> bf16 flat convert
// ---------------------------------------------------------------------------
__global__ void k_cvt(const float* __restrict__ src, __bf16* __restrict__ dst, size_t n) {
  size_t i = (size_t)blockIdx.x * blockDim.x + threadIdx.x;
  size_t stride = (size_t)gridDim.x * blockDim.x;
  for (; i < n; i += stride) dst[i] = (__bf16)src[i];
}

// f32 [b, D, Q] -> bf16 [b, Q, D]  (n-major layout for WMMA B operands)
__global__ void k_cvtT(const float* __restrict__ src, __bf16* __restrict__ dst, int D, int Q) {
  size_t base = (size_t)blockIdx.z * D * Q;
  size_t n = (size_t)D * Q;
  size_t i = (size_t)blockIdx.x * blockDim.x + threadIdx.x;
  size_t stride = (size_t)gridDim.x * blockDim.x;
  for (; i < n; i += stride) {
    int d = (int)(i / Q), q = (int)(i % Q);
    dst[base + (size_t)q * D + d] = (__bf16)src[base + i];
  }
}

// ---------------------------------------------------------------------------
// w_heads = W_qkv x z + u1ss ; split into q(+rw/rr bias), kT, v (bf16)
//   q_rw/q_rr : [b,h,i,d]  (A-operand layout for score GEMMs)
//   kT        : [b,h,j,d]  (n-major B-operand for AC)
//   v         : [b,h,d,j]  (j-contiguous B-operand for PV)
// ---------------------------------------------------------------------------
__global__ __launch_bounds__(128)
void k_qkv(const __bf16* __restrict__ Wq, const __bf16* __restrict__ zT,
           const float* __restrict__ u1ss, const float* __restrict__ rwb,
           const float* __restrict__ rrb,
           __bf16* __restrict__ q_rw, __bf16* __restrict__ q_rr,
           __bf16* __restrict__ kT, __bf16* __restrict__ vv) {
  __shared__ __attribute__((aligned(16))) __bf16 sB[2][64 * 32];
  const int b = blockIdx.y;
  const int mt = blockIdx.x >> 5;                     // 24 m-tiles of 128
  const int nt4 = blockIdx.x & 31;                    // 32 n-tiles of 64
  const int wave = threadIdx.x >> 5;
  const int m0 = mt * 128 + wave * 32, n0 = nt4 * 64;
  v8f acc[2][4] = {};
  gemm_async_128x64(Wq + (size_t)m0 * DMODEL,
                    zT + (size_t)b * QLEN * DMODEL + (size_t)n0 * DMODEL, sB, acc);
  const int lane = threadIdx.x & 31, half = lane >> 4, ml = lane & 15;
#pragma unroll
  for (int mi = 0; mi < 2; ++mi) {
#pragma unroll
    for (int nt = 0; nt < 4; ++nt) {
#pragma unroll
      for (int r = 0; r < 8; ++r) {
        int o = m0 + mi * 16 + r + half * 8;
        int q = n0 + nt * 16 + ml;
        float w = acc[mi][nt][r] + u1ss[((size_t)b * 3 * DMODEL + o) * QLEN + q];
        int blk = o >> 10;               // 0=q, 1=k, 2=v
        int oo = o & (DMODEL - 1);
        int h = oo >> 6, d = oo & 63;
        size_t bh = (size_t)(b * NHEAD + h);
        if (blk == 0) {
          q_rw[(bh * QLEN + q) * DHEAD + d] = (__bf16)(w + rwb[oo]);
          q_rr[(bh * QLEN + q) * DHEAD + d] = (__bf16)(w + rrb[oo]);
        } else if (blk == 1) {
          kT[(bh * QLEN + q) * DHEAD + d] = (__bf16)w;
        } else {
          vv[(bh * DHEAD + d) * QLEN + q] = (__bf16)w;
        }
      }
    }
  }
}

// r_head_k = W_r x pos_emb ; stored r_kT[h, c, d] (n-major B-operand for BD raw GEMM)
__global__ __launch_bounds__(128)
void k_rk(const __bf16* __restrict__ Wr, const __bf16* __restrict__ posT,
          __bf16* __restrict__ r_kT) {
  __shared__ __attribute__((aligned(16))) __bf16 sB[2][64 * 32];
  const int mt = blockIdx.x >> 5;                     // 8 m-tiles of 128
  const int nt4 = blockIdx.x & 31;
  const int wave = threadIdx.x >> 5;
  const int m0 = mt * 128 + wave * 32, n0 = nt4 * 64;
  v8f acc[2][4] = {};
  gemm_async_128x64(Wr + (size_t)m0 * DMODEL, posT + (size_t)n0 * DMODEL, sB, acc);
  const int lane = threadIdx.x & 31, half = lane >> 4, ml = lane & 15;
#pragma unroll
  for (int mi = 0; mi < 2; ++mi) {
#pragma unroll
    for (int nt = 0; nt < 4; ++nt) {
#pragma unroll
      for (int r = 0; r < 8; ++r) {
        int o = m0 + mi * 16 + r + half * 8;
        int c = n0 + nt * 16 + ml;
        int h = o >> 6, d = o & 63;
        r_kT[((size_t)h * QLEN + c) * DHEAD + d] = (__bf16)acc[mi][nt][r];
      }
    }
  }
}

// ---------------------------------------------------------------------------
// Fused attention: per (b, head, 16-row i-tile), one wave.
// AC via WMMA; BD via 2 adjacent raw tiles (WMMA) + per-row lane realign
// (rel-shift: BD[i,j] = raw[i, 2047 + j - i]); flash-style online softmax;
// PV via WMMA with P passed through LDS for C->A layout conversion.
// ---------------------------------------------------------------------------
__global__ __launch_bounds__(32)
void k_attn(const __bf16* __restrict__ q_rw, const __bf16* __restrict__ q_rr,
            const __bf16* __restrict__ kT, const __bf16* __restrict__ vv,
            const __bf16* __restrict__ r_kT, __bf16* __restrict__ av) {
  const int i0 = blockIdx.x * 16;
  const int h  = blockIdx.y;
  const int b  = blockIdx.z;
  const int lane = threadIdx.x & 31, half = lane >> 4, ml = lane & 15;
  const size_t bh = (size_t)(b * NHEAD + h);
  const __bf16* Qrw = q_rw + bh * QLEN * DHEAD + (size_t)i0 * DHEAD;
  const __bf16* Qrr = q_rr + bh * QLEN * DHEAD + (size_t)i0 * DHEAD;
  const __bf16* Kb  = kT + bh * QLEN * DHEAD;
  const __bf16* Vb  = vv + bh * DHEAD * QLEN;
  const __bf16* Rb  = r_kT + (size_t)h * QLEN * DHEAD;
  __shared__ __attribute__((aligned(16))) __bf16 sp[16 * 32];

  // Q operands are invariant across the j loop: load once.
  ABf arw[2], arr[2];
#pragma unroll
  for (int kk2 = 0; kk2 < 2; ++kk2) {
    const __bf16* pw = Qrw + ml * DHEAD + kk2 * 32 + half * 16;
    const __bf16* pr = Qrr + ml * DHEAD + kk2 * 32 + half * 16;
    arw[kk2].u[0] = *(const uint4*)(pw);
    arw[kk2].u[1] = *(const uint4*)(pw + 8);
    arr[kk2].u[0] = *(const uint4*)(pr);
    arr[kk2].u[1] = *(const uint4*)(pr + 8);
  }

  v8f O[4] = {};
  float mrun[8], lrun[8];
#pragma unroll
  for (int r = 0; r < 8; ++r) { mrun[r] = -3.0e38f; lrun[r] = 0.0f; }

  int jstart = i0 - (LOCAL - 1); if (jstart < 0) jstart = 0;
  jstart &= ~31;

  for (int jt = jstart; jt <= i0 + 15; jt += 32) {
    float s[2][8];
#pragma unroll
    for (int sub = 0; sub < 2; ++sub) {
      const int j0 = jt + sub * 16;
      // ---- AC tile ----
      v8f ac = {};
      const int jr = min(j0 + ml, QLEN - 1);
      const __bf16* kp = Kb + (size_t)jr * DHEAD;
#pragma unroll
      for (int kk2 = 0; kk2 < 2; ++kk2) {
        ABf bb;
        const __bf16* bp = kp + kk2 * 32 + half * 16;
        bb.u[0] = *(const uint4*)(bp);
        bb.u[1] = *(const uint4*)(bp + 8);
        ac = wmma_bf16(arw[kk2].v, bb.v, ac);
      }
      // ---- BD: raw tiles at columns c0..c0+31, c0 = 2032 + j0 - i0 ----
      const int c0 = (QLEN - 16) + j0 - i0;
      v8f t0 = {}, t1 = {};
      const int c0r = min(c0 + ml, QLEN - 1);
      const int c1r = min(c0 + 16 + ml, QLEN - 1);
      const __bf16* r0p = Rb + (size_t)c0r * DHEAD;
      const __bf16* r1p = Rb + (size_t)c1r * DHEAD;
#pragma unroll
      for (int kk2 = 0; kk2 < 2; ++kk2) {
        ABf b0, b1;
        const __bf16* p0 = r0p + kk2 * 32 + half * 16;
        const __bf16* p1 = r1p + kk2 * 32 + half * 16;
        b0.u[0] = *(const uint4*)(p0); b0.u[1] = *(const uint4*)(p0 + 8);
        b1.u[0] = *(const uint4*)(p1); b1.u[1] = *(const uint4*)(p1 + 8);
        t0 = wmma_bf16(arr[kk2].v, b0.v, t0);
        t1 = wmma_bf16(arr[kk2].v, b1.v, t1);
      }
      // realign raw -> BD : element (m,n) needs raw col (15 + n - m) of the pair
#pragma unroll
      for (int r = 0; r < 8; ++r) {
        const int m = r + half * 8;
        const int col = 15 + ml - m;                 // in [0,30]
        const int src = (lane & 16) | (col & 15);
        float v0 = __shfl(t0[r], src, 32);
        float v1 = __shfl(t1[r], src, 32);
        float bd = (col < 16) ? v0 : v1;
        const int i = i0 + m, j = j0 + ml;
        float sv = (ac[r] + bd) * 0.125f;            // 1/sqrt(64)
        bool masked = (j > i) || (j + LOCAL <= i);
        s[sub][r] = masked ? -3.0e38f : sv;
      }
    }
    // ---- online softmax over the 32 new columns ----
    float fac[8];
#pragma unroll
    for (int r = 0; r < 8; ++r) {
      float t = fmaxf(s[0][r], s[1][r]);
#pragma unroll
      for (int off = 1; off < 16; off <<= 1) t = fmaxf(t, __shfl_xor(t, off, 32));
      float mnew = fmaxf(mrun[r], t);
      float f = __expf(mrun[r] - mnew);
      mrun[r] = mnew;
      float p0 = __expf(s[0][r] - mnew);
      float p1 = __expf(s[1][r] - mnew);
      s[0][r] = p0; s[1][r] = p1;
      float rs = p0 + p1;
#pragma unroll
      for (int off = 1; off < 16; off <<= 1) rs += __shfl_xor(rs, off, 32);
      lrun[r] = lrun[r] * f + rs;
      fac[r] = f;
    }
#pragma unroll
    for (int dc = 0; dc < 4; ++dc)
#pragma unroll
      for (int r = 0; r < 8; ++r) O[dc][r] *= fac[r];

    // ---- P (C layout) -> LDS -> A layout ----
#pragma unroll
    for (int r = 0; r < 8; ++r) {
      const int m = r + half * 8;
      sp[m * 32 + ml]      = (__bf16)s[0][r];
      sp[m * 32 + 16 + ml] = (__bf16)s[1][r];
    }
    __syncthreads();
    ABf pa;
    const __bf16* lp = (const __bf16*)sp + ml * 32 + half * 16;
    pa.u[0] = *(const uint4*)(lp);
    pa.u[1] = *(const uint4*)(lp + 8);
    __syncthreads();

    // ---- O += P x V ----
    const int jc = min(jt + half * 16, QLEN - 16);
#pragma unroll
    for (int dc = 0; dc < 4; ++dc) {
      ABf vb;
      const __bf16* vp = Vb + (size_t)(dc * 16 + ml) * QLEN + jc;
      vb.u[0] = *(const uint4*)(vp);
      vb.u[1] = *(const uint4*)(vp + 8);
      O[dc] = wmma_bf16(pa.v, vb.v, O[dc]);
    }
  }

  // normalize and store attn_vec as av[b][i, h*64+d] (n-major for W_o GEMM)
  __bf16* avb = av + (size_t)b * QLEN * DMODEL;
#pragma unroll
  for (int r = 0; r < 8; ++r) {
    const int i = i0 + r + half * 8;
    const float inv = 1.0f / lrun[r];
#pragma unroll
    for (int dc = 0; dc < 4; ++dc) {
      const int d = h * DHEAD + dc * 16 + ml;
      avb[(size_t)i * DMODEL + d] = (__bf16)(O[dc][r] * inv);
    }
  }
}

// attn_out = W_o x attn_vec + b_o + z1ss  (pre-LN, f32)
__global__ __launch_bounds__(128)
void k_out(const __bf16* __restrict__ Wo, const __bf16* __restrict__ av,
           const float* __restrict__ b_o, const float* __restrict__ z,
           float* __restrict__ xpre) {
  __shared__ __attribute__((aligned(16))) __bf16 sB[2][64 * 32];
  const int b = blockIdx.y;
  const int mt = blockIdx.x >> 5;                     // 8 m-tiles of 128
  const int nt4 = blockIdx.x & 31;
  const int wave = threadIdx.x >> 5;
  const int m0 = mt * 128 + wave * 32, n0 = nt4 * 64;
  v8f acc[2][4] = {};
  gemm_async_128x64(Wo + (size_t)m0 * DMODEL,
                    av + (size_t)b * QLEN * DMODEL + (size_t)n0 * DMODEL, sB, acc);
  const int lane = threadIdx.x & 31, half = lane >> 4, ml = lane & 15;
#pragma unroll
  for (int mi = 0; mi < 2; ++mi) {
#pragma unroll
    for (int nt = 0; nt < 4; ++nt) {
#pragma unroll
      for (int r = 0; r < 8; ++r) {
        int o = m0 + mi * 16 + r + half * 8;
        int q = n0 + nt * 16 + ml;
        size_t idx = ((size_t)b * DMODEL + o) * QLEN + q;
        xpre[idx] = acc[mi][nt][r] + b_o[o] + z[idx];
      }
    }
  }
}

// LayerNorm over d_model per (b, q); two coalesced passes.
__global__ __launch_bounds__(256)
void k_ln(const float* __restrict__ xpre, float* __restrict__ out) {
  const int b = blockIdx.y;
  const int ql = threadIdx.x & 31;
  const int q = blockIdx.x * 32 + ql;
  const int og = threadIdx.x >> 5;                    // 8 o-groups
  const float* xp = xpre + (size_t)b * DMODEL * QLEN;
  float s = 0.f, s2 = 0.f;
  for (int o = og; o < DMODEL; o += 8) {
    float v = xp[(size_t)o * QLEN + q];
    s += v; s2 += v * v;
  }
  __shared__ float sh[2][8][32];
  sh[0][og][ql] = s; sh[1][og][ql] = s2;
  __syncthreads();
  if (og == 0) {
    float ts = 0.f, ts2 = 0.f;
#pragma unroll
    for (int r = 0; r < 8; ++r) { ts += sh[0][r][ql]; ts2 += sh[1][r][ql]; }
    float mu = ts * (1.0f / DMODEL);
    float var = ts2 * (1.0f / DMODEL) - mu * mu;
    sh[0][0][ql] = mu;
    sh[1][0][ql] = rsqrtf(var + 1e-5f);
  }
  __syncthreads();
  const float mu = sh[0][0][ql], rstd = sh[1][0][ql];
  float* op = out + (size_t)b * DMODEL * QLEN;
  for (int o = og; o < DMODEL; o += 8) {
    size_t idx = (size_t)o * QLEN + q;
    op[idx] = (xp[idx] - mu) * rstd;
  }
}

// ---------------------------------------------------------------------------
extern "C" void kernel_launch(void* const* d_in, const int* in_sizes, int n_in,
                              void* d_out, int out_size, void* d_ws, size_t ws_size,
                              hipStream_t stream) {
  const float* z1ss   = (const float*)d_in[0];
  const float* pos    = (const float*)d_in[1];
  const float* u1ss   = (const float*)d_in[2];
  const float* W_qkv  = (const float*)d_in[3];
  const float* W_r    = (const float*)d_in[4];
  const float* rwb    = (const float*)d_in[5];
  const float* rrb    = (const float*)d_in[6];
  const float* W_o    = (const float*)d_in[7];
  const float* b_o    = (const float*)d_in[8];
  float* out = (float*)d_out;

  char* w = (char*)d_ws;
  size_t off = 0;
  auto carve = [&](size_t bytes) { char* p = w + off; off += (bytes + 255) & ~(size_t)255; return p; };
  __bf16* Wq_bf  = (__bf16*)carve((size_t)3 * DMODEL * DMODEL * 2);
  __bf16* Wr_bf  = (__bf16*)carve((size_t)DMODEL * DMODEL * 2);
  __bf16* Wo_bf  = (__bf16*)carve((size_t)DMODEL * DMODEL * 2);
  __bf16* zT     = (__bf16*)carve((size_t)BSZ * QLEN * DMODEL * 2);
  __bf16* posT   = (__bf16*)carve((size_t)QLEN * DMODEL * 2);
  __bf16* q_rw   = (__bf16*)carve((size_t)BSZ * NHEAD * QLEN * DHEAD * 2);
  __bf16* q_rr   = (__bf16*)carve((size_t)BSZ * NHEAD * QLEN * DHEAD * 2);
  __bf16* kT     = (__bf16*)carve((size_t)BSZ * NHEAD * QLEN * DHEAD * 2);
  __bf16* vv     = (__bf16*)carve((size_t)BSZ * NHEAD * QLEN * DHEAD * 2);
  __bf16* r_kT   = (__bf16*)carve((size_t)NHEAD * QLEN * DHEAD * 2);
  __bf16* av     = (__bf16*)carve((size_t)BSZ * QLEN * DMODEL * 2);
  float*  xpre   = (float*)carve((size_t)BSZ * DMODEL * QLEN * 4);
  (void)ws_size; (void)n_in; (void)in_sizes; (void)out_size;

  // 1) precision conversions (weights flat, activations transposed to n-major)
  k_cvt<<<2048, 256, 0, stream>>>(W_qkv, Wq_bf, (size_t)3 * DMODEL * DMODEL);
  k_cvt<<<1024, 256, 0, stream>>>(W_r, Wr_bf, (size_t)DMODEL * DMODEL);
  k_cvt<<<1024, 256, 0, stream>>>(W_o, Wo_bf, (size_t)DMODEL * DMODEL);
  k_cvtT<<<dim3(2048, 1, BSZ), 256, 0, stream>>>(z1ss, zT, DMODEL, QLEN);
  k_cvtT<<<dim3(2048, 1, 1), 256, 0, stream>>>(pos, posT, DMODEL, QLEN);

  // 2) QKV projection GEMM (+u1ss, +rel biases) -> q_rw/q_rr/kT/v
  k_qkv<<<dim3(768, BSZ), 128, 0, stream>>>(Wq_bf, zT, u1ss, rwb, rrb, q_rw, q_rr, kT, vv);

  // 3) positional key GEMM -> r_kT
  k_rk<<<256, 128, 0, stream>>>(Wr_bf, posT, r_kT);

  // 4) fused rel-shift attention (AC+BD scores, online softmax, PV) -> av
  k_attn<<<dim3(QLEN / 16, NHEAD, BSZ), 32, 0, stream>>>(q_rw, q_rr, kT, vv, r_kT, av);

  // 5) output projection + bias + residual -> xpre
  k_out<<<dim3(256, BSZ), 128, 0, stream>>>(Wo_bf, av, b_o, z1ss, xpre);

  // 6) LayerNorm over d_model -> d_out
  k_ln<<<dim3(QLEN / 32, BSZ), 256, 0, stream>>>(xpre, out);
}
